// DSConv_54597624267059
// MI455X (gfx1250) — matrix operand order repaired and checked by reading, not compiled, not run on hardware
//
#include <hip/hip_runtime.h>
#include <hip/hip_bf16.h>

// ---------------------------------------------------------------------------
// Dynamic Snake Conv forward for MI455X (gfx1250, wave32, WMMA bf16).
// Stages:
//   P0: pack f (NCHW f32) -> padded NHWC bf16 [4][136][136][64]
//   P1: pack offset_w -> [81][32][64] bf16 (o padded 18->32), convx/convy_w -> [9][64][64]
//   K1: offset conv via v_wmma_f32_16x16x32_bf16 + bias + BN-stat atomics
//   K2: BN + tanh + cumulative offsets -> int32 gather rows idx2/idx3
//   K3: fused gather + (v2*Wx + v3*Wy) dual GEMM via WMMA -> out (NCHW f32)
// ---------------------------------------------------------------------------

typedef unsigned short u16;
typedef __attribute__((ext_vector_type(16))) __bf16 v16bf;
typedef __attribute__((ext_vector_type(8)))  __bf16 v8bf;
typedef __attribute__((ext_vector_type(8)))  float  v8f;

#define NB    4
#define NC    64
#define NH    128
#define NW    128
#define KK    9
#define HP    136                 // 128 + 2*4 pad
#define NPIX  (NB * NH * NW)      // 65536
#define NFLAT (NB * HP * HP)      // 73984 rows of 64 channels
#define OFFC  18

static __device__ __forceinline__ unsigned short f2bf(float x) {
  unsigned int u = __float_as_uint(x);
  unsigned int r = u + 0x7FFFu + ((u >> 16) & 1u);   // round-to-nearest-even
  return (unsigned short)(r >> 16);
}

static __device__ __forceinline__ v16bf mk16(const u16* p0, const u16* p1) {
  v8bf lo = *(const v8bf*)p0;     // 16B aligned by construction
  v8bf hi = *(const v8bf*)p1;
  v16bf r;
#pragma unroll
  for (int i = 0; i < 8; ++i) { r[i] = lo[i]; r[i + 8] = hi[i]; }
  return r;
}

#define WMMA_BF16(A, Bm, Cm) \
  __builtin_amdgcn_wmma_f32_16x16x32_bf16(false, (A), false, (Bm), (short)0, (Cm), false, false)

// ---------------- P0: pack f -> padded NHWC bf16 ----------------------------
__global__ void dsc_pack_f(const float* __restrict__ f, u16* __restrict__ fp) {
  int row = blockIdx.x * blockDim.x + threadIdx.x;   // 73984 rows
  if (row >= NFLAT) return;
  int b  = row / (HP * HP);
  int r2 = row - b * (HP * HP);
  int yy = r2 / HP;
  int xx = r2 - yy * HP;
  u16* dst = fp + (long)row * NC;
  if (yy < 4 || yy >= HP - 4 || xx < 4 || xx >= HP - 4) {
#pragma unroll 8
    for (int c = 0; c < NC; ++c) dst[c] = 0;
    return;
  }
  const float* src = f + (((long)b * NC) * NH + (yy - 4)) * NW + (xx - 4);
#pragma unroll 8
  for (int c = 0; c < NC; ++c) dst[c] = f2bf(src[(long)c * NH * NW]);
}

// ---------------- P1: pack weights ------------------------------------------
// woff: [81][32][64], wx/wy: [9][64][64]
__global__ void dsc_pack_w(const float* __restrict__ offw,
                           const float* __restrict__ cxw,
                           const float* __restrict__ cyw,
                           u16* __restrict__ woff, u16* __restrict__ wx, u16* __restrict__ wy) {
  int tid = blockIdx.x * blockDim.x + threadIdx.x;
  const int NWOFF = 81 * 32 * 64;       // 165888
  const int NWXY  = 9 * 64 * 64;        // 36864
  if (tid < NWOFF) {
    int kk = tid >> 11;                 // /2048
    int r  = tid & 2047;
    int o  = r >> 6, c = r & 63;
    int ky = kk / 9, kx = kk - ky * 9;
    woff[tid] = (o < OFFC) ? f2bf(offw[(((o * 64 + c) * 9 + ky) * 9) + kx]) : (u16)0;
  } else if (tid < NWOFF + NWXY) {
    int j = tid - NWOFF;
    int k = j >> 12;                    // /4096
    int r = j & 4095;
    int o = r >> 6, c = r & 63;
    wx[j] = f2bf(cxw[(o * 64 + c) * 9 + k]);     // convx_w (64,64,9,1)
  } else if (tid < NWOFF + 2 * NWXY) {
    int j = tid - NWOFF - NWXY;
    int k = j >> 12;
    int r = j & 4095;
    int o = r >> 6, c = r & 63;
    wy[j] = f2bf(cyw[(o * 64 + c) * 9 + k]);     // convy_w (64,64,1,9)
  }
}

// ---------------- K1: offset conv (implicit GEMM, WMMA) ---------------------
// 4096 tiles of 16 pixels; 64 blocks x 8 waves x 8 tiles.
__global__ void dsc_offset_conv(const u16* __restrict__ fp, const u16* __restrict__ woff,
                                const float* __restrict__ ob,
                                float* __restrict__ offws, float* __restrict__ stats) {
  const int lane = threadIdx.x & 31;
  const int wid  = threadIdx.x >> 5;
  const int lm   = lane & 15;
  const int hiA  = (lane >> 4) << 3;   // A-frag K bias: 0 / 8
  const int hiB  = (lane >> 4) << 3;   // B-frag K bias: 0 / 8
  const int hiM  = (lane >> 4) << 3;   // D rows bias:   0 / 8

  const float bias0 = ob[lm];
  const float bias1 = (lm < 2) ? ob[16 + lm] : 0.0f;

  float s0 = 0.f, q0 = 0.f, s1 = 0.f, q1 = 0.f;

  int tile0 = (blockIdx.x * 8 + wid) * 8;
  for (int t = tile0; t < tile0 + 8; ++t) {
    int pbase = t << 4;
    int b   = pbase >> 14;
    int rem = pbase & 16383;
    int yi  = rem >> 7;
    int xi0 = rem & 127;

    v8f acc0 = {0.f, 0.f, 0.f, 0.f, 0.f, 0.f, 0.f, 0.f};
    v8f acc1 = {0.f, 0.f, 0.f, 0.f, 0.f, 0.f, 0.f, 0.f};

    for (int ky = 0; ky < KK; ++ky) {
      const u16* frow = fp + ((long)(b * HP + yi + ky) * HP) * NC;
      for (int kx = 0; kx < KK; ++kx) {
        const u16* fa = frow + (xi0 + lm + kx) * NC;
        const u16* wb = woff + (long)(ky * 9 + kx) * 32 * 64;
#pragma unroll
        for (int ch = 0; ch < 2; ++ch) {
          int cb = ch << 5;
          v16bf a  = mk16(fa + cb + hiA, fa + cb + hiA + 16);
          const u16* w0 = wb + lm * 64 + cb + hiB;
          v16bf b0 = mk16(w0, w0 + 16);
          acc0 = WMMA_BF16(a, b0, acc0);
          const u16* w1 = wb + (16 + lm) * 64 + cb + hiB;
          v16bf b1 = mk16(w1, w1 + 16);
          acc1 = WMMA_BF16(a, b1, acc1);
        }
      }
    }

    // epilogue: bias, BN partial stats, store off (layout [ch][pixel])
    {
      float v[8];
#pragma unroll
      for (int e = 0; e < 8; ++e) { v[e] = acc0[e] + bias0; s0 += v[e]; q0 += v[e] * v[e]; }
      float* op = offws + (long)lm * NPIX + pbase + hiM;
      *(float4*)op       = make_float4(v[0], v[1], v[2], v[3]);
      *(float4*)(op + 4) = make_float4(v[4], v[5], v[6], v[7]);
    }
    if (lm < 2) {
      float v[8];
#pragma unroll
      for (int e = 0; e < 8; ++e) { v[e] = acc1[e] + bias1; s1 += v[e]; q1 += v[e] * v[e]; }
      float* op = offws + (long)(16 + lm) * NPIX + pbase + hiM;
      *(float4*)op       = make_float4(v[0], v[1], v[2], v[3]);
      *(float4*)(op + 4) = make_float4(v[4], v[5], v[6], v[7]);
    }
  }

  atomicAdd(&stats[lm], s0);
  atomicAdd(&stats[18 + lm], q0);
  if (lm < 2) {
    atomicAdd(&stats[16 + lm], s1);
    atomicAdd(&stats[34 + lm], q1);
  }
}

// ---------------- K2: BN + tanh + cum -> gather rows ------------------------
__global__ void dsc_bn_idx(const float* __restrict__ offws, const float* __restrict__ stats,
                           const float* __restrict__ gamma, const float* __restrict__ beta,
                           int* __restrict__ idx2, int* __restrict__ idx3) {
  int p = blockIdx.x * blockDim.x + threadIdx.x;
  if (p >= NPIX) return;
  int b   = p >> 14;
  int rem = p & 16383;
  int yi  = rem >> 7;
  int xi  = rem & 127;

  float t[OFFC];
  const float inv_n = 1.0f / (float)NPIX;
#pragma unroll
  for (int ch = 0; ch < OFFC; ++ch) {
    float mu  = stats[ch] * inv_n;
    float var = stats[18 + ch] * inv_n - mu * mu;
    float v   = offws[(long)ch * NPIX + p];
    float xh  = (v - mu) * rsqrtf(var + 1e-5f);
    t[ch] = tanhf(xh * gamma[ch] + beta[ch]);
  }

  float cz[9], cw[9];
  cz[0] = 0.f; cz[1] = t[1]; cz[2] = t[1] + t[2]; cz[3] = cz[2] + t[3];
  cz[7] = t[7]; cz[6] = t[6] + cz[7]; cz[5] = t[5] + cz[6]; cz[8] = 0.f;
  cz[4] = 0.5f * (cz[3] + cz[5]);
  cw[0] = 0.f; cw[1] = t[10]; cw[2] = t[10] + t[11]; cw[3] = cw[2] + t[12];
  cw[7] = t[16]; cw[6] = t[15] + cw[7]; cw[5] = t[14] + cw[6]; cw[8] = 0.f;
  cw[4] = 0.5f * (cw[3] + cw[5]);

#pragma unroll
  for (int k = 0; k < KK; ++k) {
    float gd = (float)(4 - k);          // g_desc = linspace(4,-4,9)
    float ga = -gd;
    float y2 = (float)(yi + 4) + gd + cz[k];
    float x2 = (float)(xi + 4) + ga + cz[k];
    float y3 = (float)(yi + 4) + ga + cw[k];
    float x3 = (float)(xi + 4) + ga - cw[k];
    int i2 = b * (NH * NW) + (int)floorf(y2) * HP + (int)floorf(x2);
    int i3 = b * (NH * NW) + (int)floorf(y3) * HP + (int)floorf(x3);
    if (i2 < 0) i2 += NFLAT;  i2 = min(max(i2, 0), NFLAT - 1);   // JAX wrap/clamp
    if (i3 < 0) i3 += NFLAT;  i3 = min(max(i3, 0), NFLAT - 1);
    idx2[p * 9 + k] = i2;
    idx3[p * 9 + k] = i3;
  }
}

// ---------------- K3: fused gather + dual GEMM (WMMA) -----------------------
// 4096 tiles; 512 blocks x 8 waves; per wave: 16 pixels x 64 out channels.
__global__ void dsc_final_conv(const u16* __restrict__ fp,
                               const u16* __restrict__ wx, const u16* __restrict__ wy,
                               const int* __restrict__ idx2, const int* __restrict__ idx3,
                               const float* __restrict__ bx, const float* __restrict__ by,
                               float* __restrict__ out) {
  const int lane = threadIdx.x & 31;
  const int wid  = threadIdx.x >> 5;
  const int lm   = lane & 15;
  const int hiA  = (lane >> 4) << 3;
  const int hiB  = (lane >> 4) << 3;
  const int hiM  = (lane >> 4) << 3;

  int tile  = blockIdx.x * 8 + wid;
  int pbase = tile << 4;
  int b   = pbase >> 14;
  int rem = pbase & 16383;
  int yi  = rem >> 7;
  int xi0 = rem & 127;
  int p   = pbase + lm;

  v8f acc[4];
#pragma unroll
  for (int nt = 0; nt < 4; ++nt) acc[nt] = (v8f){0.f,0.f,0.f,0.f,0.f,0.f,0.f,0.f};

  for (int part = 0; part < 2; ++part) {
    const int* idx  = part ? idx3 : idx2;
    const u16* wsel = part ? wy : wx;
    for (int k = 0; k < KK; ++k) {
      int row = idx[p * 9 + k];
      const u16* fa = fp + (long)row * NC;
      const u16* wb = wsel + (long)k * 64 * 64;
#pragma unroll
      for (int ch = 0; ch < 2; ++ch) {
        int cb = ch << 5;
        v16bf a = mk16(fa + cb + hiA, fa + cb + hiA + 16);
#pragma unroll
        for (int nt = 0; nt < 4; ++nt) {
          const u16* wp = wb + (nt * 16 + lm) * 64 + cb + hiB;
          v16bf bm = mk16(wp, wp + 16);
          acc[nt] = WMMA_BF16(a, bm, acc[nt]);
        }
      }
    }
  }

#pragma unroll
  for (int nt = 0; nt < 4; ++nt) {
    int o = nt * 16 + lm;
    float bias = bx[o] + by[o];
    float* op = out + (((long)(b * 64 + o) * NH + yi) * NW) + xi0 + hiM;
    *(float4*)op       = make_float4(acc[nt][0] + bias, acc[nt][1] + bias,
                                     acc[nt][2] + bias, acc[nt][3] + bias);
    *(float4*)(op + 4) = make_float4(acc[nt][4] + bias, acc[nt][5] + bias,
                                     acc[nt][6] + bias, acc[nt][7] + bias);
  }
}

// ---------------------------------------------------------------------------
extern "C" void kernel_launch(void* const* d_in, const int* in_sizes, int n_in,
                              void* d_out, int out_size, void* d_ws, size_t ws_size,
                              hipStream_t stream) {
  const float* f     = (const float*)d_in[0];
  const float* offw  = (const float*)d_in[1];
  const float* offb  = (const float*)d_in[2];
  const float* gamma = (const float*)d_in[3];
  const float* beta  = (const float*)d_in[4];
  const float* cxw   = (const float*)d_in[5];
  const float* cxb   = (const float*)d_in[6];
  const float* cyw   = (const float*)d_in[7];
  const float* cyb   = (const float*)d_in[8];
  float* out = (float*)d_out;

  // workspace carve-up (all 256B aligned)
  char* ws = (char*)d_ws;
  size_t off = 0;
  auto carve = [&](size_t bytes) { char* p = ws + off; off += (bytes + 255) & ~(size_t)255; return p; };
  u16*   fp_b   = (u16*)carve((size_t)NFLAT * NC * 2);      // 9.47 MB
  u16*   woff_b = (u16*)carve((size_t)81 * 32 * 64 * 2);    // 332 KB
  u16*   wx_b   = (u16*)carve((size_t)9 * 64 * 64 * 2);     // 74 KB
  u16*   wy_b   = (u16*)carve((size_t)9 * 64 * 64 * 2);     // 74 KB
  float* offws  = (float*)carve((size_t)OFFC * NPIX * 4);   // 4.72 MB
  float* stats  = (float*)carve(36 * 4);                    // sums / sumsq
  int*   idx2   = (int*)carve((size_t)NPIX * 9 * 4);        // 2.36 MB
  int*   idx3   = (int*)carve((size_t)NPIX * 9 * 4);        // 2.36 MB

  hipMemsetAsync(stats, 0, 256, stream);

  dsc_pack_f<<<(NFLAT + 255) / 256, 256, 0, stream>>>(f, fp_b);

  const int npk = 81 * 32 * 64 + 2 * (9 * 64 * 64);
  dsc_pack_w<<<(npk + 255) / 256, 256, 0, stream>>>(offw, cxw, cyw, woff_b, wx_b, wy_b);

  dsc_offset_conv<<<64, 256, 0, stream>>>(fp_b, woff_b, offb, offws, stats);

  dsc_bn_idx<<<NPIX / 256, 256, 0, stream>>>(offws, stats, gamma, beta, idx2, idx3);

  dsc_final_conv<<<512, 256, 0, stream>>>(fp_b, wx_b, wy_b, idx2, idx3, cxb, cyb, out);
}
